// GraphTransformerLayer_72773925863879
// MI455X (gfx1250) — matrix operand order repaired and checked by reading, not compiled, not run on hardware
//
#include <hip/hip_runtime.h>

typedef __bf16 bf16;
typedef __attribute__((ext_vector_type(16))) __bf16 v16bf;
typedef __attribute__((ext_vector_type(8)))  __bf16 v8bf;
typedef __attribute__((ext_vector_type(4)))  __bf16 v4bf;
typedef __attribute__((ext_vector_type(8)))  float  v8f;

#define B_  4
#define N_  2048
#define D_  512
#define H_  8
#define HD_ 64

static __device__ __forceinline__ v8f wmma_bf16(v16bf a, v16bf b, v8f c) {
  // D = A(16x32 bf16) * B(32x16 bf16) + C(16x16 f32)
  return __builtin_amdgcn_wmma_f32_16x16x32_bf16(false, a, false, b, (short)0, c,
                                                 false, false);
}

static __device__ __forceinline__ v8f vzero8() {
  v8f z;
#pragma unroll
  for (int i = 0; i < 8; ++i) z[i] = 0.0f;
  return z;
}

static __device__ __forceinline__ v4bf cvt4(float4 x) {
  v4bf o;
  o[0] = (bf16)x.x; o[1] = (bf16)x.y; o[2] = (bf16)x.z; o[3] = (bf16)x.w;
  return o;
}

// ---------------------------------------------------------------------------
// GEMM tiling: 128(M) x 64(N) per workgroup (128 threads / 4 waves), K-step 32.
// Wave w owns rows [32w, 32w+32): 2 A-fragments, 8 f32 accumulators,
// 8 WMMAs per K-step (each B fragment reused for both A fragments).
// Staging uses compile-time trip counts (blockDim == 128) with a load phase
// (all b128 loads outstanding) separated from the convert/store phase.
// ---------------------------------------------------------------------------

// Kernel 1: QKV projections.  Y = bf16(X @ W + b), one weight per blockIdx.z.
__global__ __launch_bounds__(128) void qkv_gemm_kernel(
    const float* __restrict__ X,
    const float* __restrict__ Wq, const float* __restrict__ bq,
    const float* __restrict__ Wk, const float* __restrict__ bk,
    const float* __restrict__ Wv, const float* __restrict__ bv,
    bf16* __restrict__ Qo, bf16* __restrict__ Ko, bf16* __restrict__ Vo) {
  const float* W; const float* bias; bf16* Y;
  if (blockIdx.z == 0)      { W = Wq; bias = bq; Y = Qo; }
  else if (blockIdx.z == 1) { W = Wk; bias = bk; Y = Ko; }
  else                      { W = Wv; bias = bv; Y = Vo; }

  __shared__ __align__(16) bf16 As[128][40];  // 128 rows x 32 K (80B rows)
  __shared__ __align__(16) bf16 Bs[32][72];   // 32 K x 64 N (144B rows)

  const int tid = threadIdx.x, wv = tid >> 5, lane = tid & 31;
  const int m0 = blockIdx.x * 128, n0 = blockIdx.y * 64;

  // Per-thread staging coordinates (compile-time strides per iteration).
  const int xr = tid >> 3, xc = (tid & 7) * 4;    // X: +16 rows / iter, 8 iters
  const int wr = tid >> 4, wc = (tid & 15) * 4;   // W: +8 rows / iter, 4 iters

  v8f acc[2][4];
#pragma unroll
  for (int s = 0; s < 2; ++s)
#pragma unroll
    for (int j = 0; j < 4; ++j) acc[s][j] = vzero8();

  for (int k0 = 0; k0 < D_; k0 += 32) {
    const float* xp = &X[(size_t)(m0 + xr) * D_ + k0 + xc];
    const float* wp = &W[(size_t)(k0 + wr) * D_ + n0 + wc];
    float4 xb[8], wb[4];
#pragma unroll
    for (int it = 0; it < 8; ++it)
      xb[it] = *(const float4*)(xp + (size_t)it * 16 * D_);
#pragma unroll
    for (int it = 0; it < 4; ++it)
      wb[it] = *(const float4*)(wp + (size_t)it * 8 * D_);
#pragma unroll
    for (int it = 0; it < 8; ++it)
      *(v4bf*)&As[xr + it * 16][xc] = cvt4(xb[it]);
#pragma unroll
    for (int it = 0; it < 4; ++it)
      *(v4bf*)&Bs[wr + it * 8][wc] = cvt4(wb[it]);
    __syncthreads();

    v16bf a[2];
#pragma unroll
    for (int s = 0; s < 2; ++s) {
      const int row = wv * 32 + s * 16 + (lane & 15);
      const int kb = (lane >> 4) * 8;
#pragma unroll
      for (int e = 0; e < 8; ++e) a[s][e] = As[row][kb + e];
#pragma unroll
      for (int e = 0; e < 8; ++e) a[s][8 + e] = As[row][kb + 16 + e];
    }
#pragma unroll
    for (int j = 0; j < 4; ++j) {
      v16bf bfr;
#pragma unroll
      for (int e = 0; e < 8; ++e) bfr[e] = Bs[(lane >> 4) * 8 + e][j * 16 + (lane & 15)];
#pragma unroll
      for (int e = 0; e < 8; ++e) bfr[8 + e] = Bs[(lane >> 4) * 8 + 16 + e][j * 16 + (lane & 15)];
#pragma unroll
      for (int s = 0; s < 2; ++s) acc[s][j] = wmma_bf16(a[s], bfr, acc[s][j]);
    }
    __syncthreads();
  }

  const int half = lane >> 4, col = lane & 15;
#pragma unroll
  for (int s = 0; s < 2; ++s) {
#pragma unroll
    for (int j = 0; j < 4; ++j) {
#pragma unroll
      for (int r = 0; r < 8; ++r) {
        const int gm = m0 + wv * 32 + s * 16 + r + 8 * half;
        const int gn = n0 + j * 16 + col;
        Y[(size_t)gm * D_ + gn] = (bf16)(acc[s][j][r] + bias[gn]);
      }
    }
  }
}

// ---------------------------------------------------------------------------
// Kernel 2: masked flash attention.
// Workgroup: 4 waves = 4 heads of one head-group, one 16-row q block.
// ---------------------------------------------------------------------------
__global__ __launch_bounds__(128) void attn_kernel(
    const bf16* __restrict__ Q, const bf16* __restrict__ K,
    const bf16* __restrict__ V, const int* __restrict__ adj,
    bf16* __restrict__ O) {
  __shared__ __align__(16) bf16 Ks[32][264];    // 32 keys x 256 cols (528B rows)
  __shared__ __align__(16) bf16 Vs[32][264];
  __shared__ __align__(16) int  adjs[16][32];   // 16 q rows x 32 keys
  __shared__ __align__(16) bf16 Ps[4][16][40];  // per-wave P scratch

  const int tid = threadIdx.x, wv = tid >> 5, lane = tid & 31;
  const int q0 = blockIdx.x * 16;
  const int hg = blockIdx.y;        // head group (0..1), 4 heads each
  const int b  = blockIdx.z;
  const int hh = hg * 4 + wv;       // global head of this wave
  const int half = lane >> 4, col = lane & 15;
  const float scale = 0.125f;       // 1/sqrt(64)

  // Staging coordinates: 32 rows x 32 chunks of 8 bf16; +4 rows / iter.
  const int sr = tid >> 5, sc = (tid & 31) * 8;
  // Prefetch coordinates: 32 rows x 4 cachelines (128B) per row.
  const int pr = tid >> 2, pc = (tid & 3) * 64;

  // Preload Q fragments (A-layout) for the full head dim (two K=32 chunks).
  v16bf qf[2];
  {
    const bf16* qp = Q + ((size_t)b * N_ + q0 + (lane & 15)) * D_ + hh * HD_;
    const int kb = half * 8;
#pragma unroll
    for (int c = 0; c < 2; ++c) {
#pragma unroll
      for (int e = 0; e < 8; ++e) qf[c][e] = qp[c * 32 + kb + e];
#pragma unroll
      for (int e = 0; e < 8; ++e) qf[c][8 + e] = qp[c * 32 + kb + 16 + e];
    }
  }

  v8f acc[4];
#pragma unroll
  for (int j = 0; j < 4; ++j) acc[j] = vzero8();
  float mrow[8], lrow[8];
#pragma unroll
  for (int r = 0; r < 8; ++r) { mrow[r] = -__builtin_inff(); lrow[r] = 0.0f; }

  for (int m0 = 0; m0 < N_; m0 += 32) {
    const bf16* kp = K + ((size_t)b * N_ + m0) * D_ + hg * 256;
    const bf16* vp = V + ((size_t)b * N_ + m0) * D_ + hg * 256;

    // Stage K/V: all 16 b128 loads outstanding, then LDS stores.
    v8bf kb_[8], vb_[8];
#pragma unroll
    for (int it = 0; it < 8; ++it) {
      kb_[it] = *(const v8bf*)&kp[(size_t)(sr + it * 4) * D_ + sc];
      vb_[it] = *(const v8bf*)&vp[(size_t)(sr + it * 4) * D_ + sc];
    }
    {
      // adj tile: one int4 per thread.
      const int ar = tid >> 3, ac = (tid & 7) * 4;
      const int4 av =
          *(const int4*)&adj[((size_t)b * N_ + q0 + ar) * N_ + m0 + ac];
      *(int4*)&adjs[ar][ac] = av;
    }
    // Prefetch next key tile while this one is consumed.
    if (m0 + 32 < N_) {
      __builtin_prefetch(&kp[(size_t)(32 + pr) * D_ + pc], 0, 0);
      __builtin_prefetch(&vp[(size_t)(32 + pr) * D_ + pc], 0, 0);
    }
#pragma unroll
    for (int it = 0; it < 8; ++it) {
      *(v8bf*)&Ks[sr + it * 4][sc] = kb_[it];
      *(v8bf*)&Vs[sr + it * 4][sc] = vb_[it];
    }
    __syncthreads();

    // Scores: S(16q x 32keys) = Q @ K^T ; two 16x16 tiles, each 2 WMMAs.
    v8f sacc[2];
    sacc[0] = vzero8(); sacc[1] = vzero8();
#pragma unroll
    for (int t = 0; t < 2; ++t) {
#pragma unroll
      for (int c = 0; c < 2; ++c) {
        v16bf bfr;  // B[kk][n] = K[key = t*16+n][hd = c*32+kk]
        const int cc = wv * 64 + c * 32 + half * 8;
#pragma unroll
        for (int e = 0; e < 8; ++e) bfr[e] = Ks[t * 16 + col][cc + e];
#pragma unroll
        for (int e = 0; e < 8; ++e) bfr[8 + e] = Ks[t * 16 + col][cc + 16 + e];
        sacc[t] = wmma_bf16(qf[c], bfr, sacc[t]);
      }
    }

    // Online softmax per row (row = r + 8*half, spread over 16-lane group).
#pragma unroll
    for (int r = 0; r < 8; ++r) {
      const int qrow = r + 8 * half;
      const int msk0 = adjs[qrow][col];
      const int msk1 = adjs[qrow][16 + col];
      const float s0 = msk0 ? sacc[0][r] * scale : -__builtin_inff();
      const float s1 = msk1 ? sacc[1][r] * scale : -__builtin_inff();
      float tm = fmaxf(s0, s1);
#pragma unroll
      for (int d = 1; d < 16; d <<= 1) tm = fmaxf(tm, __shfl_xor(tm, d, 32));
      const float mn = fmaxf(mrow[r], tm);
      const float alpha =
          (mn == -__builtin_inff()) ? 1.0f : __expf(mrow[r] - mn);
      const float p0 = msk0 ? __expf(s0 - mn) : 0.0f;
      const float p1 = msk1 ? __expf(s1 - mn) : 0.0f;
      float ps = p0 + p1;
#pragma unroll
      for (int d = 1; d < 16; d <<= 1) ps += __shfl_xor(ps, d, 32);
      lrow[r] = lrow[r] * alpha + ps;
      mrow[r] = mn;
#pragma unroll
      for (int j = 0; j < 4; ++j) acc[j][r] *= alpha;
      Ps[wv][qrow][col]      = (bf16)p0;
      Ps[wv][qrow][16 + col] = (bf16)p1;
    }
    // Per-wave LDS round-trip (C-layout -> A-layout); DS ops are in-order,
    // make the dependency explicit before the A-fragment reload.
    asm volatile("s_wait_dscnt 0" ::: "memory");

    v16bf pf;
    {
      const int row = lane & 15, kb = half * 8;
#pragma unroll
      for (int e = 0; e < 8; ++e) pf[e] = Ps[wv][row][kb + e];
#pragma unroll
      for (int e = 0; e < 8; ++e) pf[8 + e] = Ps[wv][row][kb + 16 + e];
    }
#pragma unroll
    for (int j = 0; j < 4; ++j) {
      v16bf vfr;  // B[key][n] = V[key][hd = j*16+n]
#pragma unroll
      for (int e = 0; e < 8; ++e) vfr[e] = Vs[half * 8 + e][wv * 64 + j * 16 + col];
#pragma unroll
      for (int e = 0; e < 8; ++e) vfr[8 + e] = Vs[half * 8 + 16 + e][wv * 64 + j * 16 + col];
      acc[j] = wmma_bf16(pf, vfr, acc[j]);
    }
    __syncthreads();
  }

#pragma unroll
  for (int r = 0; r < 8; ++r) {
    const float inv = 1.0f / lrow[r];
    const int gm = q0 + r + 8 * half;
#pragma unroll
    for (int j = 0; j < 4; ++j)
      O[((size_t)b * N_ + gm) * D_ + hh * HD_ + j * 16 + col] =
          (bf16)(acc[j][r] * inv);
  }
}

// ---------------------------------------------------------------------------
// Kernel 3: output projection + residual.  Y = f32(H @ Wo + bo + X)
// ---------------------------------------------------------------------------
__global__ __launch_bounds__(128) void oproj_gemm_kernel(
    const bf16* __restrict__ A, const float* __restrict__ W,
    const float* __restrict__ bias, const float* __restrict__ X,
    float* __restrict__ Y) {
  __shared__ __align__(16) bf16 As[128][40];
  __shared__ __align__(16) bf16 Bs[32][72];

  const int tid = threadIdx.x, wv = tid >> 5, lane = tid & 31;
  const int m0 = blockIdx.x * 128, n0 = blockIdx.y * 64;

  const int hr = tid >> 2, hc = (tid & 3) * 8;    // H: +32 rows / iter, 4 iters
  const int wr = tid >> 4, wc = (tid & 15) * 4;   // W: +8 rows / iter, 4 iters

  v8f acc[2][4];
#pragma unroll
  for (int s = 0; s < 2; ++s)
#pragma unroll
    for (int j = 0; j < 4; ++j) acc[s][j] = vzero8();

  for (int k0 = 0; k0 < D_; k0 += 32) {
    const bf16* ap = &A[(size_t)(m0 + hr) * D_ + k0 + hc];
    const float* wp = &W[(size_t)(k0 + wr) * D_ + n0 + wc];
    v8bf ab[4]; float4 wb[4];
#pragma unroll
    for (int it = 0; it < 4; ++it)
      ab[it] = *(const v8bf*)(ap + (size_t)it * 32 * D_);
#pragma unroll
    for (int it = 0; it < 4; ++it)
      wb[it] = *(const float4*)(wp + (size_t)it * 8 * D_);
#pragma unroll
    for (int it = 0; it < 4; ++it)
      *(v8bf*)&As[hr + it * 32][hc] = ab[it];
#pragma unroll
    for (int it = 0; it < 4; ++it)
      *(v4bf*)&Bs[wr + it * 8][wc] = cvt4(wb[it]);
    __syncthreads();

    v16bf a[2];
#pragma unroll
    for (int s = 0; s < 2; ++s) {
      const int row = wv * 32 + s * 16 + (lane & 15);
      const int kb = (lane >> 4) * 8;
#pragma unroll
      for (int e = 0; e < 8; ++e) a[s][e] = As[row][kb + e];
#pragma unroll
      for (int e = 0; e < 8; ++e) a[s][8 + e] = As[row][kb + 16 + e];
    }
#pragma unroll
    for (int j = 0; j < 4; ++j) {
      v16bf bfr;
#pragma unroll
      for (int e = 0; e < 8; ++e) bfr[e] = Bs[(lane >> 4) * 8 + e][j * 16 + (lane & 15)];
#pragma unroll
      for (int e = 0; e < 8; ++e) bfr[8 + e] = Bs[(lane >> 4) * 8 + 16 + e][j * 16 + (lane & 15)];
#pragma unroll
      for (int s = 0; s < 2; ++s) acc[s][j] = wmma_bf16(a[s], bfr, acc[s][j]);
    }
    __syncthreads();
  }

  const int half = lane >> 4, col = lane & 15;
#pragma unroll
  for (int s = 0; s < 2; ++s) {
#pragma unroll
    for (int j = 0; j < 4; ++j) {
#pragma unroll
      for (int r = 0; r < 8; ++r) {
        const int gm = m0 + wv * 32 + s * 16 + r + 8 * half;
        const int gn = n0 + j * 16 + col;
        Y[(size_t)gm * D_ + gn] =
            acc[s][j][r] + bias[gn] + X[(size_t)gm * D_ + gn];
      }
    }
  }
}

// ---------------------------------------------------------------------------
// Kernel 4: LayerNorm over D=512.  One wave per row; float4 IO.
// ---------------------------------------------------------------------------
__global__ __launch_bounds__(256) void ln_kernel(
    const float* __restrict__ Y, const float* __restrict__ gamma,
    const float* __restrict__ beta, float* __restrict__ out) {
  const int row = blockIdx.x * 8 + (threadIdx.x >> 5);
  const int lane = threadIdx.x & 31;
  const float* yr = Y + (size_t)row * D_;
  float4 v[4];
  float s = 0.0f, ss = 0.0f;
#pragma unroll
  for (int i = 0; i < 4; ++i) {
    v[i] = *(const float4*)&yr[lane * 4 + i * 128];
    s += v[i].x + v[i].y + v[i].z + v[i].w;
    ss += v[i].x * v[i].x + v[i].y * v[i].y + v[i].z * v[i].z + v[i].w * v[i].w;
  }
#pragma unroll
  for (int d = 1; d < 32; d <<= 1) {
    s += __shfl_xor(s, d, 32);
    ss += __shfl_xor(ss, d, 32);
  }
  const float mu = s * (1.0f / D_);
  const float var = ss * (1.0f / D_) - mu * mu;
  const float rstd = rsqrtf(var + 1e-5f);
#pragma unroll
  for (int i = 0; i < 4; ++i) {
    const int c = lane * 4 + i * 128;
    const float4 g = *(const float4*)&gamma[c];
    const float4 be = *(const float4*)&beta[c];
    float4 o;
    o.x = (v[i].x - mu) * rstd * g.x + be.x;
    o.y = (v[i].y - mu) * rstd * g.y + be.y;
    o.z = (v[i].z - mu) * rstd * g.z + be.z;
    o.w = (v[i].w - mu) * rstd * g.w + be.w;
    *(float4*)&out[(size_t)row * D_ + c] = o;
  }
}

// ---------------------------------------------------------------------------
extern "C" void kernel_launch(void* const* d_in, const int* in_sizes, int n_in,
                              void* d_out, int out_size, void* d_ws,
                              size_t ws_size, hipStream_t stream) {
  (void)in_sizes; (void)n_in; (void)out_size; (void)ws_size;
  const float* x     = (const float*)d_in[0];
  const int*   adj   = (const int*)d_in[1];
  const float* Wq    = (const float*)d_in[2];
  const float* bq    = (const float*)d_in[3];
  const float* Wk    = (const float*)d_in[4];
  const float* bk    = (const float*)d_in[5];
  const float* Wv    = (const float*)d_in[6];
  const float* bv    = (const float*)d_in[7];
  const float* Wo    = (const float*)d_in[8];
  const float* bo    = (const float*)d_in[9];
  const float* gamma = (const float*)d_in[10];
  const float* beta  = (const float*)d_in[11];
  float* out = (float*)d_out;

  const size_t MD = (size_t)B_ * N_ * D_;  // 4 Mi elements
  bf16* Qb = (bf16*)d_ws;
  bf16* Kb = Qb + MD;
  bf16* Vb = Kb + MD;
  bf16* Hb = Vb + MD;
  float* Yf = (float*)(Hb + MD);

  dim3 g1(B_ * N_ / 128, D_ / 64, 3);
  qkv_gemm_kernel<<<g1, 128, 0, stream>>>(x, Wq, bq, Wk, bk, Wv, bv, Qb, Kb, Vb);

  dim3 g2(N_ / 16, H_ / 4, B_);
  attn_kernel<<<g2, 128, 0, stream>>>(Qb, Kb, Vb, adj, Hb);

  dim3 g3(B_ * N_ / 128, D_ / 64);
  oproj_gemm_kernel<<<g3, 128, 0, stream>>>(Hb, Wo, bo, x, Yf);

  ln_kernel<<<B_ * N_ / 8, 256, 0, stream>>>(Yf, gamma, beta, out);
}
